// MoE_30837865185522
// MI455X (gfx1250) — matrix operand order repaired and checked by reading, not compiled, not run on hardware
//
#include <hip/hip_runtime.h>
#include <hip/hip_bf16.h>
#include <math.h>

// Problem constants (from reference): B=32768, D=1200, E=16, H=1024, O=1, K=4
#define B_ROWS 32768
#define D_DIM  1200
#define DP_DIM 1280        // D padded to 5 chunks x 8 steps x 32 (zero-filled)
#define KSTEPS (DP_DIM / 32)   // 40
#define KCHUNK 8               // K-steps staged per barrier
#define E_NUM  16
#define H_DIM  1024
#define LN_EPSF 1e-5f

typedef __attribute__((ext_vector_type(16))) __bf16 v16bf;
typedef __attribute__((ext_vector_type(8)))  float  v8f;

union Frag16 { v16bf v; unsigned short u[16]; unsigned long long q[4]; };

__device__ __forceinline__ unsigned short f32_to_bf16_u(float f) {
    unsigned int u = __float_as_uint(f);
    u += 0x7FFFu + ((u >> 16) & 1u);   // round-to-nearest-even
    return (unsigned short)(u >> 16);
}

// ---------------------------------------------------------------- init
__global__ void moe_init(int* __restrict__ cnt) {
    if (threadIdx.x < E_NUM) cnt[threadIdx.x] = 0;
}

// ---------------------------------------------------------------- W1 -> bf16, transposed [E][H][DP]
__global__ __launch_bounds__(256) void moe_prep(const float* __restrict__ W1,
                                                unsigned short* __restrict__ W1T) {
    int bid = blockIdx.x;
    int e = bid / DP_DIM;
    int k = bid % DP_DIM;
    int n0 = threadIdx.x * 4;
    float4 w = make_float4(0.f, 0.f, 0.f, 0.f);
    if (k < D_DIM)
        w = *reinterpret_cast<const float4*>(W1 + ((size_t)e * D_DIM + k) * H_DIM + n0);
    size_t base = ((size_t)e * H_DIM + n0) * DP_DIM + k;
    W1T[base]              = f32_to_bf16_u(w.x);
    W1T[base + DP_DIM]     = f32_to_bf16_u(w.y);
    W1T[base + 2 * DP_DIM] = f32_to_bf16_u(w.z);
    W1T[base + 3 * DP_DIM] = f32_to_bf16_u(w.w);
}

// ---------------------------------------------------------------- gating
__global__ __launch_bounds__(256) void moe_gate(
    const float* __restrict__ x, const float* __restrict__ wg,
    int* __restrict__ cnt, int* __restrict__ rlist, float* __restrict__ gval,
    float* __restrict__ part_imp)
{
    __shared__ float logits[16][16];
    __shared__ int   topi[16][4];
    __shared__ float topg[16][4];

    int tid = threadIdx.x;
    int r   = tid >> 4;
    int e   = tid & 15;
    int row = blockIdx.x * 16 + r;

    const float* xr = x + (size_t)row * D_DIM;
    float acc = 0.f;
    #pragma unroll 4
    for (int d = 0; d < D_DIM; ++d) acc += xr[d] * wg[d * E_NUM + e];
    logits[r][e] = acc;
    __syncthreads();

    if (e == 0) {
        float l[16];
        float mx = -1e30f;
        for (int i = 0; i < 16; ++i) { l[i] = logits[r][i]; mx = fmaxf(mx, l[i]); }
        float s = 0.f;
        for (int i = 0; i < 16; ++i) { l[i] = expf(l[i] - mx); s += l[i]; }
        float inv = 1.f / s;
        for (int i = 0; i < 16; ++i) l[i] *= inv;

        int used = 0; int ti[4]; float tv[4]; float gs = 1e-6f;
        for (int k = 0; k < 4; ++k) {
            int bi = -1; float bv = -1.f;
            for (int i = 0; i < 16; ++i)
                if (!((used >> i) & 1) && l[i] > bv) { bv = l[i]; bi = i; }
            used |= 1 << bi; ti[k] = bi; tv[k] = bv; gs += bv;
        }
        float rgs = 1.f / gs;
        for (int k = 0; k < 4; ++k) {
            float g = tv[k] * rgs;
            topi[r][k] = ti[k]; topg[r][k] = g;
            int pos = atomicAdd(&cnt[ti[k]], 1);
            rlist[(size_t)ti[k] * B_ROWS + pos] = row | (k << 20);
            gval [(size_t)ti[k] * B_ROWS + pos] = g;
        }
    }
    __syncthreads();

    if (tid < 16) {   // deterministic per-block importance partials
        float s = 0.f;
        for (int rr = 0; rr < 16; ++rr)
            for (int k = 0; k < 4; ++k)
                if (topi[rr][k] == tid) s += topg[rr][k];
        part_imp[blockIdx.x * 16 + tid] = s;
    }
}

// ---------------------------------------------------------------- expert WMMA
// Block = 256 thr / 8 waves; 32 gathered rows x H=1024; K staged in 256-wide
// chunks (8 WMMA K-steps per barrier). Each wave: 128 cols, 2x8 accumulators.
__global__ __launch_bounds__(256) void moe_expert(
    const float* __restrict__ x,  const unsigned short* __restrict__ W1T,
    const float* __restrict__ b1, const float* __restrict__ gma,
    const float* __restrict__ bta, const float* __restrict__ W2,
    const float* __restrict__ b2, const int* __restrict__ cnt,
    const int* __restrict__ rlist, const float* __restrict__ gval,
    float* __restrict__ contrib)
{
    // Permuted A staging: [step s][sub*32 + lane][frag elem i] -> lane reads
    // 32 contiguous bytes per sub-tile per step.
    __shared__ unsigned short AtileP[KCHUNK * 64 * 16];   // 16 KB
    __shared__ int   srow[32];
    __shared__ int   sslot[32];
    __shared__ float sgate[32];
    __shared__ float red_sum[32], red_sq[32], red_dot[32];

    const int tilesPerExpert = B_ROWS / 32;            // 1024
    int e    = blockIdx.x / tilesPerExpert;
    int tile = blockIdx.x % tilesPerExpert;
    int n_e  = cnt[e];
    if (tile * 32 >= n_e) return;                      // block-uniform exit

    int tid  = threadIdx.x;
    int wave = tid >> 5, lane = tid & 31;
    int half = lane >> 4, l15 = lane & 15;
    int nbase = wave * 128;

    if (tid < 32) {
        int slot = tile * 32 + tid;
        if (slot < n_e) {
            int ent = rlist[(size_t)e * B_ROWS + slot];
            srow[tid]  = ent & 0xFFFFF;
            sslot[tid] = ent >> 20;
            sgate[tid] = gval[(size_t)e * B_ROWS + slot];
        } else { srow[tid] = -1; sslot[tid] = 0; sgate[tid] = 0.f; }
        red_sum[tid] = 0.f; red_sq[tid] = 0.f; red_dot[tid] = 0.f;
    }
    __syncthreads();

    // Staging assignment: per step, 1024 elems (32 rows x 32 k) / 256 thr.
    const int mrow = tid >> 3;            // 0..31 gathered row
    const int kk   = (tid & 7) * 4;       // 4-aligned k within a 32-step
    const int myrow = srow[mrow];
    const float* xrow = x + (size_t)(myrow < 0 ? 0 : myrow) * D_DIM;
    const int halfa = kk >> 4;
    const int ia    = kk & 15;
    const int slotbase = ((mrow >> 4) * 32 + halfa * 16 + (mrow & 15)) * 16 + ia;

    v8f acc0[8], acc1[8];
    #pragma unroll
    for (int j = 0; j < 8; ++j)
        #pragma unroll
        for (int rr = 0; rr < 8; ++rr) { acc0[j][rr] = 0.f; acc1[j][rr] = 0.f; }

    const unsigned short* Bbase = W1T + (size_t)e * H_DIM * DP_DIM;

    for (int ktb = 0; ktb < KSTEPS; ktb += KCHUNK) {   // 5 chunks
        __syncthreads();
        // stage KCHUNK steps of A (zero-pad k >= 1200; 4 | 1200 so no tail)
        #pragma unroll
        for (int s = 0; s < KCHUNK; ++s) {
            int kg = (ktb + s) * 32 + kk;
            float4 a = make_float4(0.f, 0.f, 0.f, 0.f);
            if (myrow >= 0 && kg < D_DIM)
                a = *reinterpret_cast<const float4*>(xrow + kg);
            unsigned long long pk =
                  (unsigned long long)f32_to_bf16_u(a.x)
                | ((unsigned long long)f32_to_bf16_u(a.y) << 16)
                | ((unsigned long long)f32_to_bf16_u(a.z) << 32)
                | ((unsigned long long)f32_to_bf16_u(a.w) << 48);
            *reinterpret_cast<unsigned long long*>(&AtileP[s * 1024 + slotbase]) = pk;
        }
        __syncthreads();

        // barrier-free WMMA over the staged chunk
        for (int s = 0; s < KCHUNK; ++s) {
            int k0 = (ktb + s) * 32;
            Frag16 af0, af1;
            af0.v = *reinterpret_cast<const v16bf*>(&AtileP[s * 1024 + lane * 16]);
            af1.v = *reinterpret_cast<const v16bf*>(&AtileP[s * 1024 + (32 + lane) * 16]);
            #pragma unroll
            for (int j = 0; j < 8; ++j) {
                const unsigned short* bp =
                    Bbase + (size_t)(nbase + j * 16 + l15) * DP_DIM + k0 + half * 16;
                Frag16 bf;
                bf.v = *reinterpret_cast<const v16bf*>(bp);
                __builtin_prefetch(bp + 32, 0, 1);   // next K line -> global_prefetch
                acc0[j] = __builtin_amdgcn_wmma_f32_16x16x32_bf16(
                              false, af0.v, false, bf.v, (short)0, acc0[j], false, false);
                acc1[j] = __builtin_amdgcn_wmma_f32_16x16x32_bf16(
                              false, af1.v, false, bf.v, (short)0, acc1[j], false, false);
            }
        }
    }

    // bias add
    #pragma unroll
    for (int j = 0; j < 8; ++j) {
        int n = nbase + j * 16 + l15;
        float bb = b1[e * H_DIM + n];
        #pragma unroll
        for (int rr = 0; rr < 8; ++rr) { acc0[j][rr] += bb; acc1[j][rr] += bb; }
    }

    // layernorm stats via LDS float atomics per row
    #pragma unroll
    for (int rr = 0; rr < 8; ++rr) {
        float s0 = 0.f, q0 = 0.f, s1 = 0.f, q1 = 0.f;
        #pragma unroll
        for (int j = 0; j < 8; ++j) {
            float v0 = acc0[j][rr]; s0 += v0; q0 += v0 * v0;
            float v1 = acc1[j][rr]; s1 += v1; q1 += v1 * v1;
        }
        int m = rr + half * 8;
        atomicAdd(&red_sum[m], s0);      atomicAdd(&red_sq[m], q0);
        atomicAdd(&red_sum[16 + m], s1); atomicAdd(&red_sq[16 + m], q1);
    }
    __syncthreads();

    float mu0[8], rs0[8], mu1[8], rs1[8];
    #pragma unroll
    for (int rr = 0; rr < 8; ++rr) {
        int m = rr + half * 8;
        float a = red_sum[m] * (1.f / H_DIM);
        float v = red_sq[m] * (1.f / H_DIM) - a * a;
        mu0[rr] = a; rs0[rr] = rsqrtf(v + LN_EPSF);
        float a1 = red_sum[16 + m] * (1.f / H_DIM);
        float v1 = red_sq[16 + m] * (1.f / H_DIM) - a1 * a1;
        mu1[rr] = a1; rs1[rr] = rsqrtf(v1 + LN_EPSF);
    }

    // normalize, gamma/beta, relu, fused dot with W2 column
    float dp0[8], dp1[8];
    #pragma unroll
    for (int rr = 0; rr < 8; ++rr) { dp0[rr] = 0.f; dp1[rr] = 0.f; }
    #pragma unroll
    for (int j = 0; j < 8; ++j) {
        int n = nbase + j * 16 + l15;
        float g  = gma[e * H_DIM + n];
        float bt = bta[e * H_DIM + n];
        float w2 = W2 [e * H_DIM + n];
        #pragma unroll
        for (int rr = 0; rr < 8; ++rr) {
            float h0 = fmaxf((acc0[j][rr] - mu0[rr]) * rs0[rr] * g + bt, 0.f);
            float h1 = fmaxf((acc1[j][rr] - mu1[rr]) * rs1[rr] * g + bt, 0.f);
            dp0[rr] += h0 * w2;
            dp1[rr] += h1 * w2;
        }
    }
    #pragma unroll
    for (int rr = 0; rr < 8; ++rr) {
        atomicAdd(&red_dot[rr + half * 8], dp0[rr]);
        atomicAdd(&red_dot[16 + rr + half * 8], dp1[rr]);
    }
    __syncthreads();

    if (tid < 32 && srow[tid] >= 0) {
        float o = red_dot[tid] + b2[e];
        o = 1.f / (1.f + expf(-o));
        contrib[(size_t)srow[tid] * 4 + sslot[tid]] = sgate[tid] * o;
    }
}

// ---------------------------------------------------------------- finalize y
__global__ void moe_finalize(const float* __restrict__ contrib, float* __restrict__ y) {
    int b = blockIdx.x * 256 + threadIdx.x;
    const float* c = contrib + (size_t)b * 4;
    y[b] = (c[0] + c[1]) + (c[2] + c[3]);
}

// ---------------------------------------------------------------- aux loss
__global__ void moe_loss(const float* __restrict__ part_imp,
                         const int* __restrict__ cnt, float* __restrict__ out_loss) {
    __shared__ float imp[16], ld[16];
    int t = threadIdx.x;
    if (t < 16) {
        float s = 0.f;
        for (int bIdx = 0; bIdx < B_ROWS / 16; ++bIdx) s += part_imp[bIdx * 16 + t];
        imp[t] = s;
        ld[t]  = (float)cnt[t];
    }
    __syncthreads();
    if (t == 0) {
        float loss = 0.f;
        for (int pass = 0; pass < 2; ++pass) {
            const float* v = pass ? ld : imp;
            float mean = 0.f;
            for (int i = 0; i < 16; ++i) mean += v[i];
            mean *= (1.f / 16.f);
            float var = 0.f;
            for (int i = 0; i < 16; ++i) { float d = v[i] - mean; var += d * d; }
            var *= (1.f / 15.f);                    // ddof = 1
            loss += var / (mean * mean + 1e-10f);
        }
        out_loss[0] = 0.5f * loss;
    }
}

// ---------------------------------------------------------------- launch
extern "C" void kernel_launch(void* const* d_in, const int* in_sizes, int n_in,
                              void* d_out, int out_size, void* d_ws, size_t ws_size,
                              hipStream_t stream) {
    const float* x   = (const float*)d_in[0];
    const float* wg  = (const float*)d_in[1];
    const float* W1  = (const float*)d_in[2];
    const float* b1  = (const float*)d_in[3];
    const float* gma = (const float*)d_in[4];
    const float* bta = (const float*)d_in[5];
    const float* W2  = (const float*)d_in[6];
    const float* b2  = (const float*)d_in[7];
    // d_in[8] = k (==4, baked in)

    char* ws = (char*)d_ws;
    size_t off = 0;
    unsigned short* W1T = (unsigned short*)(ws + off);
    off += (size_t)E_NUM * H_DIM * DP_DIM * 2;                 // 41.9 MB
    int*   cnt      = (int*)(ws + off);   off += 256;
    int*   rlist    = (int*)(ws + off);   off += (size_t)E_NUM * B_ROWS * 4;
    float* gvalv    = (float*)(ws + off); off += (size_t)E_NUM * B_ROWS * 4;
    float* contrib  = (float*)(ws + off); off += (size_t)B_ROWS * 4 * 4;
    float* part_imp = (float*)(ws + off); off += (size_t)(B_ROWS / 16) * 16 * 4;

    float* y     = (float*)d_out;
    float* lossp = y + B_ROWS;

    moe_init    <<<1, 16, 0, stream>>>(cnt);
    moe_prep    <<<E_NUM * DP_DIM, 256, 0, stream>>>(W1, W1T);
    moe_gate    <<<B_ROWS / 16, 256, 0, stream>>>(x, wg, cnt, rlist, gvalv, part_imp);
    moe_expert  <<<E_NUM * (B_ROWS / 32), 256, 0, stream>>>(x, W1T, b1, gma, bta, W2, b2,
                                                            cnt, rlist, gvalv, contrib);
    moe_finalize<<<B_ROWS / 256, 256, 0, stream>>>(contrib, y);
    moe_loss    <<<1, 32, 0, stream>>>(part_imp, cnt, lossp);
}
